// MHATest_11871289606770
// MI455X (gfx1250) — matrix operand order repaired and checked by reading, not compiled, not run on hardware
//
#include <hip/hip_runtime.h>
#include <hip/hip_bf16.h>
#include <cstdint>

typedef __bf16 bf16;
typedef __attribute__((ext_vector_type(8)))  __bf16 v8bf;
typedef __attribute__((ext_vector_type(16))) __bf16 v16bf;
typedef __attribute__((ext_vector_type(8)))  float  v8f;

#ifndef __has_builtin
#define __has_builtin(x) 0
#endif
#if __has_builtin(__builtin_amdgcn_global_load_async_to_lds_b128)
#define ASYNC_LDS 1
#else
#define ASYNC_LDS 0
#endif

// vector/addrspace types matching the async-load builtin signature:
//   void (v4i addrspace(1)*, v4i addrspace(3)*, imm i32 offset, imm i32 cpol)
typedef int v4i_vs __attribute__((vector_size(16)));
typedef __attribute__((address_space(1))) v4i_vs* gptr_v4i;
typedef __attribute__((address_space(3))) v4i_vs* lptr_v4i;

// ---------------------------------------------------------------------------
// WMMA fragment helpers (CDNA5 wave32, V_WMMA_F32_16X16X32_BF16)
//
// A (16x32 bf16, row-major source): lane l holds row m=l&15; k-chunks at
//   ko=(l>>4)*8 : VGPR0..3 <- k=[ko,ko+8), VGPR4..7 <- k=[ko+16,ko+24)
//   => two contiguous 16-byte loads per lane.
// B (32x16 bf16) loaded from an [n,k] (i.e. B-transposed row-major) source:
//   lane l holds col n=l&15; same k order as A with kk=(l>>4)*8:
//   k=[kk,kk+8) then k=[kk+16,kk+24) => two contiguous 16-byte loads per lane.
// C/D (16x16 f32): lane l col n=l&15, VGPR r row m = r + 8*(l>>4).
// ---------------------------------------------------------------------------
static __device__ __forceinline__ v16bf make_a(const bf16* p0, const bf16* p1) {
  v8bf lo = *(const v8bf*)p0;
  v8bf hi = *(const v8bf*)p1;
  v16bf r;
#pragma unroll
  for (int i = 0; i < 8; ++i) { r[i] = lo[i]; r[i + 8] = hi[i]; }
  return r;
}
static __device__ __forceinline__ v16bf make_a_f32(const float* p0, const float* p1) {
  v8f lo = *(const v8f*)p0;
  v8f hi = *(const v8f*)p1;
  v16bf r;
#pragma unroll
  for (int i = 0; i < 8; ++i) { r[i] = (bf16)lo[i]; r[i + 8] = (bf16)hi[i]; }
  return r;
}
static __device__ __forceinline__ v16bf make_b(const bf16* p) { return make_a(p, p + 16); }
static __device__ __forceinline__ v8f vzero8() {
  v8f z;
#pragma unroll
  for (int i = 0; i < 8; ++i) z[i] = 0.f;
  return z;
}
static __device__ __forceinline__ v8f wmma_bf16(v16bf a, v16bf b, v8f c) {
  return __builtin_amdgcn_wmma_f32_16x16x32_bf16(false, a, false, b, (short)0, c,
                                                 false, false);
}

// ---------------------------------------------------------------------------
// Async global -> LDS 16-byte copy (CDNA5 GLOBAL_LOAD_ASYNC_TO_LDS_B128,
// ASYNCcnt-tracked). Fallback: plain load + ds_store (same barrier protocol).
// ---------------------------------------------------------------------------
static __device__ __forceinline__ void cp16_g2l(const bf16* g, bf16* l) {
#if ASYNC_LDS
  __builtin_amdgcn_global_load_async_to_lds_b128((gptr_v4i)(void*)g, (lptr_v4i)(void*)l,
                                                 0, 0);
#else
  *(v8bf*)l = *(const v8bf*)g;
#endif
}
template <int N>
static __device__ __forceinline__ void async_wait() {
#if ASYNC_LDS
  asm volatile("s_wait_asynccnt %0" ::"i"(N) : "memory");
#endif
}

// ---------------------------------------------------------------------------
// f32 -> bf16 conversion
// ---------------------------------------------------------------------------
__global__ void cvt_bf16_kernel(const float* __restrict__ s, bf16* __restrict__ d,
                                long n) {
  long i = (long)blockIdx.x * blockDim.x + threadIdx.x;
  if (i < n) d[i] = (bf16)s[i];
}

// ---------------------------------------------------------------------------
// conv1: x[16,3,96,512] * w1[32,3,k,3] (pad width 1) -> h[16,32,wd,512] (+bf16)
// grid (4, wd, 16), block 128; weights staged in LDS, 32 f32 accumulators.
// ---------------------------------------------------------------------------
__global__ __launch_bounds__(128)
void conv1_kernel(const float* __restrict__ x, const float* __restrict__ w1,
                  const float* __restrict__ b1, float* __restrict__ h,
                  bf16* __restrict__ hbf, int k, int wd) {
  __shared__ float sw[32 * 3 * 10 * 3];
  const int nw = 32 * 3 * k * 3;
  for (int j = threadIdx.x; j < nw; j += 128) sw[j] = w1[j];
  __syncthreads();
  const int d = blockIdx.x * 128 + threadIdx.x;
  const int w = blockIdx.y;
  const int b = blockIdx.z;
  const int w9k = 9 * k;
  float acc[32];
#pragma unroll
  for (int oc = 0; oc < 32; ++oc) acc[oc] = b1[oc];
  for (int ic = 0; ic < 3; ++ic)
    for (int dh = 0; dh < k; ++dh) {
      const float* xr = x + ((size_t)(b * 3 + ic) * 96 + (w + dh)) * 512;
#pragma unroll
      for (int dw = 0; dw < 3; ++dw) {
        int dd = d + dw - 1;
        float xv = (dd >= 0 && dd < 512) ? xr[dd] : 0.f;
        int widx = (ic * k + dh) * 3 + dw;
#pragma unroll
        for (int oc = 0; oc < 32; ++oc)
          acc[oc] = fmaf(xv, sw[oc * w9k + widx], acc[oc]);
      }
    }
#pragma unroll
  for (int oc = 0; oc < 32; ++oc) {
    size_t o = ((size_t)(b * 32 + oc) * wd + w) * 512 + d;
    h[o] = acc[oc];
    hbf[o] = (bf16)acc[oc];
  }
}

// ---------------------------------------------------------------------------
// GEMM: out[M,512] = A[M,512] * Bt[512,512]^T + bias   (Bt is [n,k] row-major,
// which is exactly the torch weight [e,d] layout). M = 512*wd (mult of 128).
// Block 256 = 8 waves; block tile 128x64, wave tile 16x64 (4 WMMA tiles).
// B tiles (64x32 bf16 = 4KB) double-buffered in LDS via async global->LDS
// copies: one B128 copy per thread per k-step, overlapped with WMMA compute.
// ---------------------------------------------------------------------------
__global__ __launch_bounds__(256)
void gemm512_kernel(const bf16* __restrict__ A, const bf16* __restrict__ Bt,
                    const float* __restrict__ bias, float* __restrict__ outF,
                    bf16* __restrict__ outB, int f32out) {
  __shared__ bf16 sB[2][64 * 32];  // [buf][n][k]
  const int tid = threadIdx.x;
  const int lane = tid & 31;
  const int wave = tid >> 5;
  const int m0 = blockIdx.x * 128 + wave * 16;
  const int n0 = blockIdx.y * 64;
  const int nn = lane & 15;
  const int ko = (lane >> 4) * 8;
  const int kk = (lane >> 4) * 8;
  const int mo = (lane >> 4) * 8;

  // stage mapping: thread t copies 8 bf16 (16B): n = t>>2, k-chunk = (t&3)*8
  const int sn = tid >> 2;
  const int sc = (tid & 3) * 8;
  const bf16* gsrc = Bt + (size_t)(n0 + sn) * 512 + sc;
  bf16* ldst0 = &sB[0][sn * 32 + sc];
  bf16* ldst1 = &sB[1][sn * 32 + sc];

  v8f acc[4];
#pragma unroll
  for (int j = 0; j < 4; ++j) acc[j] = vzero8();
  const bf16* arow = A + (size_t)(m0 + nn) * 512;

  cp16_g2l(gsrc, ldst0);  // prologue: stage k0 = 0 into buf 0
  int buf = 0;
  for (int k0 = 0; k0 < 512; k0 += 32) {
    const bool more = (k0 + 32) < 512;
    if (more) cp16_g2l(gsrc + k0 + 32, buf ? ldst0 : ldst1);
    if (more) async_wait<1>(); else async_wait<0>();
    __syncthreads();  // all waves waited on their own copies -> tile resident
    if (more) __builtin_prefetch((const void*)(arow + k0 + 64), 0, 3);
    v16bf a = make_a(arow + k0 + ko, arow + k0 + ko + 16);
    const bf16* bb = &sB[buf][0];
#pragma unroll
    for (int j = 0; j < 4; ++j)
      acc[j] = wmma_bf16(a, make_b(bb + (j * 16 + nn) * 32 + kk), acc[j]);
    __syncthreads();  // done reading buf before anyone overwrites it
    buf ^= 1;
  }
#pragma unroll
  for (int j = 0; j < 4; ++j) {
    int n = n0 + j * 16 + nn;
    float bv = bias[n];
#pragma unroll
    for (int r = 0; r < 8; ++r) {
      size_t o = (size_t)(m0 + mo + r) * 512 + n;
      float v = acc[j][r] + bv;
      if (f32out) outF[o] = v; else outB[o] = (bf16)v;
    }
  }
}

// ---------------------------------------------------------------------------
// Attention for one group g (4096 groups): S = QK^T/8, softmax, O = P*V.
// Gather: flat row R=g*wd+q  ->  c=R%32, w=(R/32)%wd, bh=(R/32)/wd.
// Scatter (block-uniform): c2=g%32, bh=g/32, w2=q.
// LDS: K tile 12KB, V^T tile 12KB, scores 36KB (f32, 96x96 padded).
// ---------------------------------------------------------------------------
__global__ __launch_bounds__(256)
void attn_kernel(const bf16* __restrict__ qbf, const bf16* __restrict__ kbf,
                 const bf16* __restrict__ vbf, bf16* __restrict__ aobf, int wd) {
  __shared__ bf16 sK[96 * 64];
  __shared__ bf16 sVt[64 * 96];
  __shared__ float sS[96 * 96];
  const int g = blockIdx.x;
  const int tid = threadIdx.x, lane = tid & 31, wave = tid >> 5;
  const int nn = lane & 15;
  const int ko = (lane >> 4) * 8;
  const int kk = (lane >> 4) * 8;
  const int mo = (lane >> 4) * 8;

  // ---- stage K rows and V^T into LDS (zero-pad invalid rows) ----
  for (int idx = tid; idx < 96 * 64; idx += 256) {
    int r = idx >> 6, e = idx & 63;
    bf16 kv = (bf16)0.f, vv = (bf16)0.f;
    if (r < wd) {
      long R = (long)g * wd + r;
      int cch = (int)(R & 31);
      long t2 = R >> 5;
      int w = (int)(t2 % wd);
      int bh = (int)(t2 / wd);
      size_t base = (((size_t)((bh >> 3) * 32 + cch) * wd + w) << 9) + ((bh & 7) << 6);
      kv = kbf[base + e];
      vv = vbf[base + e];
    }
    sK[idx] = kv;
    sVt[e * 96 + r] = vv;
  }
  __syncthreads();

  // ---- S = Q K^T * (1/8) : 6x6 tiles over 8 waves, K-depth 64 = 2 steps ----
  for (int t = wave; t < 36; t += 8) {
    int mt = t / 6, nt = t % 6;
    int qpos = mt * 16 + nn;
    if (qpos > wd - 1) qpos = wd - 1;       // clamp: padded rows discarded later
    long R = (long)g * wd + qpos;
    int cch = (int)(R & 31);
    long t2 = R >> 5;
    int w = (int)(t2 % wd);
    int bh = (int)(t2 / wd);
    const bf16* qrow =
        qbf + (((size_t)((bh >> 3) * 32 + cch) * wd + w) << 9) + ((bh & 7) << 6);
    const bf16* krow = sK + (nt * 16 + nn) * 64 + kk;
    v8f acc = vzero8();
#pragma unroll
    for (int k0 = 0; k0 < 64; k0 += 32)
      acc = wmma_bf16(make_a(qrow + k0 + ko, qrow + k0 + ko + 16),
                      make_b(krow + k0), acc);
#pragma unroll
    for (int r = 0; r < 8; ++r)
      sS[(mt * 16 + mo + r) * 96 + nt * 16 + nn] = acc[r] * 0.125f;
  }
  __syncthreads();

  // ---- row softmax over valid columns; zero padded columns ----
  for (int r = wave; r < wd; r += 8) {
    float* row = sS + r * 96;
    float mx = -3.0e38f;
    for (int j = lane; j < wd; j += 32) mx = fmaxf(mx, row[j]);
#pragma unroll
    for (int off = 16; off; off >>= 1) mx = fmaxf(mx, __shfl_xor(mx, off));
    float sm = 0.f;
    for (int j = lane; j < wd; j += 32) {
      float e = __expf(row[j] - mx);
      row[j] = e;
      sm += e;
    }
#pragma unroll
    for (int off = 16; off; off >>= 1) sm += __shfl_xor(sm, off);
    float inv = 1.f / sm;
    for (int j = lane; j < wd; j += 32) row[j] *= inv;
    for (int j = wd + lane; j < 96; j += 32) row[j] = 0.f;
  }
  __syncthreads();

  // ---- O = P V : 6x4 tiles over 8 waves, K-depth 96 = 3 steps ----
  const int c2 = g & 31, bh2 = g >> 5;
  const int h2 = bh2 & 7;
  const size_t dbase = ((size_t)((bh2 >> 3) * 32 + c2) * wd) << 9;
  for (int t = wave; t < 24; t += 8) {
    int mt = t >> 2, nt = t & 3;
    const float* srow = sS + (mt * 16 + nn) * 96;
    const bf16* vrow = sVt + (nt * 16 + nn) * 96 + kk;
    v8f acc = vzero8();
#pragma unroll
    for (int k0 = 0; k0 < 96; k0 += 32)
      acc = wmma_bf16(make_a_f32(srow + k0 + ko, srow + k0 + ko + 16),
                      make_b(vrow + k0), acc);
#pragma unroll
    for (int r = 0; r < 8; ++r) {
      int row = mt * 16 + mo + r;
      if (row < wd)
        aobf[dbase + (size_t)row * 512 + h2 * 64 + nt * 16 + nn] = (bf16)acc[r];
    }
  }
}

// ---------------------------------------------------------------------------
// SE: channel means, tiny MLP 32->8->32 with sigmoid, then h += ao^2 * y.
// ---------------------------------------------------------------------------
__global__ __launch_bounds__(256)
void mean_kernel(const float* __restrict__ ao, float* __restrict__ ym, int wd) {
  __shared__ float red[256];
  long n = (long)wd * 512;
  const float* p = ao + (size_t)blockIdx.x * n;
  float s = 0.f;
  for (long j = threadIdx.x; j < n; j += 256) s += p[j];
  red[threadIdx.x] = s;
  __syncthreads();
  for (int o = 128; o; o >>= 1) {
    if ((int)threadIdx.x < o) red[threadIdx.x] += red[threadIdx.x + o];
    __syncthreads();
  }
  if (threadIdx.x == 0) ym[blockIdx.x] = red[0] / (float)n;
}

__global__ __launch_bounds__(512)
void se_kernel(const float* __restrict__ ym, const float* __restrict__ w1,
               const float* __restrict__ w2, float* __restrict__ y) {
  __shared__ float z[128];
  int t = threadIdx.x;
  if (t < 128) {
    int b = t >> 3, j = t & 7;
    float s = 0.f;
    for (int c = 0; c < 32; ++c) s += ym[b * 32 + c] * w1[j * 32 + c];
    z[t] = fmaxf(s, 0.f);
  }
  __syncthreads();
  int b = t >> 5, c = t & 31;
  float s = 0.f;
#pragma unroll
  for (int j = 0; j < 8; ++j) s += z[b * 8 + j] * w2[c * 8 + j];
  y[t] = 1.f / (1.f + __expf(-s));
}

__global__ void se_apply_kernel(float* __restrict__ h, const float* __restrict__ ao,
                                const float* __restrict__ y, bf16* __restrict__ hbf,
                                int wd) {
  long i = (long)blockIdx.x * 256 + threadIdx.x;
  long n = (long)16 * 32 * wd * 512;
  if (i < n) {
    int bc = (int)(i / ((long)wd * 512));
    float a = ao[i];
    float v = h[i] + a * a * y[bc];
    h[i] = v;
    hbf[i] = (bf16)v;
  }
}

// ---------------------------------------------------------------------------
// conv2 as implicit GEMM on WMMA: rows m = b*wd2 + w' (M2 = 16*wd2), cols oc.
// K loop over ic(32) x kh(k) x kw(512). Grid wd2 blocks x 64 threads (2 waves,
// one 16x16 oc-tile each).
// ---------------------------------------------------------------------------
__global__ __launch_bounds__(64)
void conv2_kernel(const bf16* __restrict__ hbf, const bf16* __restrict__ w2,
                  const float* __restrict__ b2, float* __restrict__ c2out,
                  int k, int wd, int wd2) {
  const int lane = threadIdx.x & 31, wave = threadIdx.x >> 5;
  const int m0 = blockIdx.x * 16;
  const int ocb = wave * 16;
  const int nn = lane & 15;
  const int ko = (lane >> 4) * 8;
  const int kk = (lane >> 4) * 8;
  const int mo = (lane >> 4) * 8;
  const int mrow = m0 + nn;
  const int b = mrow / wd2, wp = mrow % wd2;
  v8f acc = vzero8();
  for (int ic = 0; ic < 32; ++ic)
    for (int dh = 0; dh < k; ++dh) {
      const bf16* arow = hbf + ((size_t)(b * 32 + ic) * wd + (wp + dh)) * 512;
      const bf16* brow =
          w2 + (((size_t)(ocb + nn) * 32 + ic) * k + dh) * 512 + kk;
      for (int d0 = 0; d0 < 512; d0 += 32)
        acc = wmma_bf16(make_a(arow + d0 + ko, arow + d0 + ko + 16),
                        make_b(brow + d0), acc);
    }
  const float bv = b2[ocb + nn];
#pragma unroll
  for (int r = 0; r < 8; ++r) {
    int m = m0 + mo + r;
    int bb = m / wd2, ww = m % wd2;
    c2out[((size_t)bb * 32 + (ocb + nn)) * wd2 + ww] = fmaxf(acc[r] + bv, 0.f);
  }
}

__global__ __launch_bounds__(512)
void maxpool_kernel(const float* __restrict__ c2o, float* __restrict__ hidden,
                    int wd2, int slot) {
  int t = threadIdx.x;  // 512 = 16 b x 32 oc
  int b = t >> 5, oc = t & 31;
  const float* p = c2o + (size_t)t * wd2;
  float m = p[0];
  for (int w = 1; w < wd2; ++w) m = fmaxf(m, p[w]);
  hidden[b * 320 + slot * 32 + oc] = m;
}

__global__ void fc_kernel(const float* __restrict__ hidden, const float* __restrict__ fcw,
                          const float* __restrict__ fcb, float* __restrict__ out) {
  int t = threadIdx.x;
  if (t >= 32) return;
  int b = t >> 1, c = t & 1;
  float s = fcb[c];
  for (int q = 0; q < 320; ++q) s += hidden[b * 320 + q] * fcw[c * 320 + q];
  out[b * 2 + c] = s;
}

// ---------------------------------------------------------------------------
// host launcher
// ---------------------------------------------------------------------------
extern "C" void kernel_launch(void* const* d_in, const int* in_sizes, int n_in,
                              void* d_out, int out_size, void* d_ws, size_t ws_size,
                              hipStream_t stream) {
  (void)in_sizes; (void)out_size; (void)ws_size;
  const float *x, *c1w[10], *c1b[10], *c2w[10], *c2b[10];
  const float *qw, *qb, *kw, *kb, *vw, *vb, *pw, *pb, *sw1, *sw2, *fcw, *fcb;
  if (n_in >= 40) {  // lists expanded into individual inputs
    int p = 0;
    x = (const float*)d_in[p++];
    for (int i = 0; i < 10; ++i) c1w[i] = (const float*)d_in[p++];
    for (int i = 0; i < 10; ++i) c1b[i] = (const float*)d_in[p++];
    for (int i = 0; i < 10; ++i) c2w[i] = (const float*)d_in[p++];
    for (int i = 0; i < 10; ++i) c2b[i] = (const float*)d_in[p++];
    qw = (const float*)d_in[p++]; qb = (const float*)d_in[p++];
    kw = (const float*)d_in[p++]; kb = (const float*)d_in[p++];
    vw = (const float*)d_in[p++]; vb = (const float*)d_in[p++];
    pw = (const float*)d_in[p++]; pb = (const float*)d_in[p++];
    sw1 = (const float*)d_in[p++]; sw2 = (const float*)d_in[p++];
    fcw = (const float*)d_in[p++]; fcb = (const float*)d_in[p++];
  } else {  // each list concatenated flat into one input
    x = (const float*)d_in[0];
    const float* a1 = (const float*)d_in[1];
    const float* a2 = (const float*)d_in[2];
    const float* a3 = (const float*)d_in[3];
    const float* a4 = (const float*)d_in[4];
    size_t o1 = 0, o3 = 0;
    for (int i = 0; i < 10; ++i) {
      int k = i + 1;
      c1w[i] = a1 + o1; o1 += (size_t)32 * 3 * k * 3;
      c1b[i] = a2 + (size_t)i * 32;
      c2w[i] = a3 + o3; o3 += (size_t)32 * 32 * k * 512;
      c2b[i] = a4 + (size_t)i * 32;
    }
    qw = (const float*)d_in[5];  qb = (const float*)d_in[6];
    kw = (const float*)d_in[7];  kb = (const float*)d_in[8];
    vw = (const float*)d_in[9];  vb = (const float*)d_in[10];
    pw = (const float*)d_in[11]; pb = (const float*)d_in[12];
    sw1 = (const float*)d_in[13]; sw2 = (const float*)d_in[14];
    fcw = (const float*)d_in[15]; fcb = (const float*)d_in[16];
  }

  // workspace carve (wd padded to 96 everywhere; buffers reused per branch)
  char* base = (char*)d_ws;
  size_t off = 0;
  auto carve = [&](size_t bytes) -> char* {
    char* p = base + off;
    off = (off + bytes + 255) & ~(size_t)255;
    return p;
  };
  const size_t EL = (size_t)16 * 32 * 96 * 512;  // 25,165,824 elements
  float* h    = (float*)carve(EL * 4);
  bf16*  hbf  = (bf16*)carve(EL * 2);
  bf16*  qm   = (bf16*)carve(EL * 2);
  bf16*  km   = (bf16*)carve(EL * 2);
  bf16*  vm   = (bf16*)carve(EL * 2);
  bf16*  aom  = (bf16*)carve(EL * 2);
  float* aof  = (float*)carve(EL * 4);
  bf16*  wqb  = (bf16*)carve((size_t)512 * 512 * 2);
  bf16*  wkb  = (bf16*)carve((size_t)512 * 512 * 2);
  bf16*  wvb  = (bf16*)carve((size_t)512 * 512 * 2);
  bf16*  wpb  = (bf16*)carve((size_t)512 * 512 * 2);
  bf16*  w2b  = (bf16*)carve((size_t)32 * 32 * 10 * 512 * 2);
  float* c2o  = (float*)carve((size_t)16 * 32 * 96 * 4);
  float* ym   = (float*)carve(16 * 32 * 4);
  float* yse  = (float*)carve(16 * 32 * 4);

  float* out_f  = (float*)d_out;
  float* hidden = out_f + 32;

  const long wn = 512 * 512;
  cvt_bf16_kernel<<<(wn + 255) / 256, 256, 0, stream>>>(qw, wqb, wn);
  cvt_bf16_kernel<<<(wn + 255) / 256, 256, 0, stream>>>(kw, wkb, wn);
  cvt_bf16_kernel<<<(wn + 255) / 256, 256, 0, stream>>>(vw, wvb, wn);
  cvt_bf16_kernel<<<(wn + 255) / 256, 256, 0, stream>>>(pw, wpb, wn);

  for (int i = 0; i < 10; ++i) {
    const int k = i + 1, wd = 96 - i, wd2 = 96 - 2 * i;
    const int M = 512 * wd;

    conv1_kernel<<<dim3(4, wd, 16), 128, 0, stream>>>(x, c1w[i], c1b[i], h, hbf, k, wd);

    const long w2n = (long)32 * 32 * k * 512;
    cvt_bf16_kernel<<<(w2n + 255) / 256, 256, 0, stream>>>(c2w[i], w2b, w2n);

    dim3 gg(M / 128, 8);
    gemm512_kernel<<<gg, 256, 0, stream>>>(hbf, wqb, qb, nullptr, qm, 0);
    gemm512_kernel<<<gg, 256, 0, stream>>>(hbf, wkb, kb, nullptr, km, 0);
    gemm512_kernel<<<gg, 256, 0, stream>>>(hbf, wvb, vb, nullptr, vm, 0);

    attn_kernel<<<4096, 256, 0, stream>>>(qm, km, vm, aom, wd);

    gemm512_kernel<<<gg, 256, 0, stream>>>(aom, wpb, pb, aof, nullptr, 1);

    mean_kernel<<<512, 256, 0, stream>>>(aof, ym, wd);
    se_kernel<<<1, 512, 0, stream>>>(ym, sw1, sw2, yse);
    const long n = (long)M * 512;
    se_apply_kernel<<<(n + 255) / 256, 256, 0, stream>>>(h, aof, yse, hbf, wd);

    conv2_kernel<<<wd2, 64, 0, stream>>>(hbf, w2b, c2b[i], c2o, k, wd, wd2);
    maxpool_kernel<<<1, 512, 0, stream>>>(c2o, hidden, wd2, i);
  }

  fc_kernel<<<1, 32, 0, stream>>>(hidden, fcw, fcb, out_f);
}
